// TransConvLayer_14877766714048
// MI455X (gfx1250) — matrix operand order repaired and checked by reading, not compiled, not run on hardware
//
#include <hip/hip_runtime.h>

typedef __attribute__((ext_vector_type(16))) __bf16          v16bf;
typedef __attribute__((ext_vector_type(8)))  float           v8f;
typedef __attribute__((ext_vector_type(8)))  unsigned short  v8us;

// round-to-nearest (ties away): 2 VALU ops instead of 4+ for full RNE
__device__ __forceinline__ unsigned short f2bf(float f) {
    return (unsigned short)((__float_as_uint(f) + 0x8000u) >> 16);
}
__device__ __forceinline__ unsigned int pk2bf(float lo, float hi) {
    unsigned int a = __float_as_uint(lo) + 0x8000u;
    unsigned int b = __float_as_uint(hi) + 0x8000u;
    return (b & 0xffff0000u) | (a >> 16);
}
__device__ __forceinline__ float bf2f(unsigned short h) {
    return __uint_as_float(((unsigned int)h) << 16);
}
__device__ __forceinline__ v8f zero8() {
    v8f z = {0.f, 0.f, 0.f, 0.f, 0.f, 0.f, 0.f, 0.f};
    return z;
}
// Build a 16xbf16 A/B fragment from two 16-byte chunks (LDS or global).
__device__ __forceinline__ v16bf ld_frag(const unsigned short* p0, const unsigned short* p1) {
    union { v8us u[2]; v16bf v; } x;
    x.u[0] = *(const v8us*)p0;
    x.u[1] = *(const v8us*)p1;
    return x.v;
}
__device__ __forceinline__ v8f wmma_bf16(v16bf a, v16bf b, v8f c) {
    return __builtin_amdgcn_wmma_f32_16x16x32_bf16(false, a, false, b, (short)0, c, false, false);
}

// ---------------------------------------------------------------- zero scratch
__global__ void k_zero(float* p, int n) {
    int i = blockIdx.x * blockDim.x + threadIdx.x;
    if (i < n) p[i] = 0.f;
}

// ------------------------------------------------- stage 0: transpose weights to bf16
__global__ void k_wt(const float* __restrict__ Wq, const float* __restrict__ Wk,
                     const float* __restrict__ Wv, unsigned short* __restrict__ Wt)
{
    int i = blockIdx.x * 256 + threadIdx.x;
    if (i >= 3 * 1024 * 256) return;
    int mat = i >> 18;
    int rem = i & 262143;
    int c = rem >> 8, k = rem & 255;
    const float* W = (mat == 0) ? Wq : (mat == 1) ? Wk : Wv;
    Wt[i] = f2bf(W[(size_t)k * 1024 + c]);
}

// ------------------------------------------------- stage 1: Q projection GEMM
__global__ void k_q(const float* __restrict__ Xq, const float* __restrict__ bq,
                    const unsigned short* __restrict__ Wt,
                    unsigned short* __restrict__ Qb,
                    float* __restrict__ scal, int N)
{
    __shared__ __align__(16) unsigned short Asl[64][40];

    const int tid  = threadIdx.x;
    const int lane = tid & 31, wid = tid >> 5;
    const int r0   = blockIdx.x * 64;
    const int c0   = blockIdx.y * 64;

    const int wy = wid >> 1, wx = wid & 1;
    const int m0 = wy * 32, n0 = wx * 32;
    const int lq = lane & 15, lh = lane >> 4;

    v8f acc[2][2];
    acc[0][0] = zero8(); acc[0][1] = zero8(); acc[1][0] = zero8(); acc[1][1] = zero8();

    for (int ks = 0; ks < 8; ++ks) {
        const int k0 = ks * 32;
        #pragma unroll
        for (int i = 0; i < 4; ++i) {
            int e  = tid + i * 128;
            int r  = e >> 3;
            int c4 = (e & 7) * 4;
            int row = r0 + r; if (row >= N) row = N - 1;
            float4 f = *(const float4*)(Xq + (size_t)row * 256 + k0 + c4);
            unsigned long long p = ((unsigned long long)pk2bf(f.z, f.w) << 32)
                                 | (unsigned long long)pk2bf(f.x, f.y);
            *(unsigned long long*)&Asl[r][c4] = p;
        }
        __syncthreads();

        v16bf a[2], b[2];
        #pragma unroll
        for (int mt = 0; mt < 2; ++mt) {
            const unsigned short* p = &Asl[m0 + mt * 16 + lq][lh * 16];
            a[mt] = ld_frag(p, p + 8);
        }
        #pragma unroll
        for (int nt = 0; nt < 2; ++nt) {
            const unsigned short* p = Wt + (size_t)(c0 + n0 + nt * 16 + lq) * 256 + k0 + lh * 16;
            b[nt] = ld_frag(p, p + 8);
        }
        #pragma unroll
        for (int mt = 0; mt < 2; ++mt)
            #pragma unroll
            for (int nt = 0; nt < 2; ++nt)
                acc[mt][nt] = wmma_bf16(a[mt], b[nt], acc[mt][nt]);
        __syncthreads();
    }

    float ss = 0.f;
    if (r0 + 64 <= N) {
        #pragma unroll
        for (int nt = 0; nt < 2; ++nt) {
            const int c = c0 + n0 + nt * 16 + lq;
            const float bias = bq[c];
            #pragma unroll
            for (int mt = 0; mt < 2; ++mt)
                #pragma unroll
                for (int r = 0; r < 8; ++r) {
                    int row = r0 + m0 + mt * 16 + lh * 8 + r;
                    float v = acc[mt][nt][r] + bias;
                    Qb[(size_t)row * 1024 + c] = f2bf(v);
                    ss += v * v;
                }
        }
    } else {
        #pragma unroll
        for (int nt = 0; nt < 2; ++nt) {
            const int c = c0 + n0 + nt * 16 + lq;
            const float bias = bq[c];
            #pragma unroll
            for (int mt = 0; mt < 2; ++mt)
                #pragma unroll
                for (int r = 0; r < 8; ++r) {
                    int row = r0 + m0 + mt * 16 + lh * 8 + r;
                    float v = acc[mt][nt][r] + bias;
                    if (row < N) {
                        Qb[(size_t)row * 1024 + c] = f2bf(v);
                        ss += v * v;
                    }
                }
        }
    }
    #pragma unroll
    for (int off = 16; off >= 1; off >>= 1) ss += __shfl_xor(ss, off, 32);
    if (lane == 0) atomicAdd(&scal[0], ss);
}

// ------------------------------------------------- stage 2 (FUSED): K/V projection + kvs[h] = K_h^T V_h
__global__ void k_kv(const float* __restrict__ Xs,
                     const float* __restrict__ bk, const float* __restrict__ bv,
                     const unsigned short* __restrict__ Wt,
                     unsigned short* __restrict__ Vb,
                     float* __restrict__ kvs, float* __restrict__ ksum,
                     float* __restrict__ scal, int N)
{
    __shared__ __align__(16) unsigned short Xl[32][264];  // [l][k] bf16
    __shared__ __align__(16) unsigned short Kt[128][40];  // [m][l]
    __shared__ __align__(16) unsigned short Vt[128][40];  // [d][l]

    const int tid  = threadIdx.x;
    const int lane = tid & 31, wid = tid >> 5;
    const int h    = blockIdx.y;
    const int CH   = (((N + 31) / 32) + 31) & ~31;
    const int lbeg = blockIdx.x * CH;
    const int lend = (lbeg + CH < N) ? (lbeg + CH) : N;

    const int mw = wid & 1, nw = wid >> 1;
    const int lq = lane & 15, lh = lane >> 4;
    const int lrow = mw * 16 + lh * 8;

    const unsigned short* Wkp = Wt + 262144;
    const unsigned short* Wvp = Wt + 524288;
    float bK[2], bV[2];
    #pragma unroll
    for (int nt = 0; nt < 2; ++nt) {
        bK[nt] = bk[h * 128 + nw * 32 + nt * 16 + lq];
        bV[nt] = bv[h * 128 + nw * 32 + nt * 16 + lq];
    }

    v8f kvacc[4][2];
    #pragma unroll
    for (int i = 0; i < 4; ++i) { kvacc[i][0] = zero8(); kvacc[i][1] = zero8(); }
    float ss = 0.f, cs[2] = {0.f, 0.f};

    for (int l0 = lbeg; l0 < lend; l0 += 32) {
        const bool fullstep = (l0 + 32 <= lend);
        if (fullstep) {
            #pragma unroll
            for (int i = 0; i < 8; ++i) {
                int e = tid + i * 256;
                int r = e >> 6, c4 = (e & 63) * 4;
                float4 f = *(const float4*)(Xs + (size_t)(l0 + r) * 256 + c4);
                *(unsigned long long*)&Xl[r][c4] =
                    ((unsigned long long)pk2bf(f.z, f.w) << 32) | (unsigned long long)pk2bf(f.x, f.y);
            }
        } else {
            #pragma unroll
            for (int i = 0; i < 8; ++i) {
                int e = tid + i * 256;
                int r = e >> 6, c4 = (e & 63) * 4;
                int row = l0 + r;
                float4 f;
                if (row < lend) f = *(const float4*)(Xs + (size_t)row * 256 + c4);
                else { f.x = 0.f; f.y = 0.f; f.z = 0.f; f.w = 0.f; }
                *(unsigned long long*)&Xl[r][c4] =
                    ((unsigned long long)pk2bf(f.z, f.w) << 32) | (unsigned long long)pk2bf(f.x, f.y);
            }
        }
        __syncthreads();

        v8f ka[2], va[2];
        ka[0] = zero8(); ka[1] = zero8(); va[0] = zero8(); va[1] = zero8();
        #pragma unroll
        for (int kk = 0; kk < 8; ++kk) {
            const int k0 = kk * 32;
            const unsigned short* pa = &Xl[mw * 16 + lq][k0 + lh * 16];
            v16bf a = ld_frag(pa, pa + 8);
            #pragma unroll
            for (int nt = 0; nt < 2; ++nt) {
                const size_t coff = (size_t)(h * 128 + nw * 32 + nt * 16 + lq) * 256 + k0 + lh * 16;
                const unsigned short* pk = Wkp + coff;
                const unsigned short* pv = Wvp + coff;
                ka[nt] = wmma_bf16(a, ld_frag(pk, pk + 8), ka[nt]);
                va[nt] = wmma_bf16(a, ld_frag(pv, pv + 8), va[nt]);
            }
        }

        if (fullstep) {
            #pragma unroll
            for (int nt = 0; nt < 2; ++nt) {
                const int c = nw * 32 + nt * 16 + lq;
                v8us kpk, vpk;
                #pragma unroll
                for (int r = 0; r < 8; ++r) {
                    float kvv = ka[nt][r] + bK[nt];
                    float vvv = va[nt][r] + bV[nt];
                    ss += kvv * kvv;
                    cs[nt] += kvv;
                    kpk[r] = f2bf(kvv);
                    vpk[r] = f2bf(vvv);
                }
                *(v8us*)&Kt[c][lrow] = kpk;
                *(v8us*)&Vt[c][lrow] = vpk;
                #pragma unroll
                for (int r = 0; r < 8; ++r)
                    Vb[(size_t)(l0 + lrow + r) * 1024 + h * 128 + c] = vpk[r];
            }
        } else {
            #pragma unroll
            for (int nt = 0; nt < 2; ++nt) {
                const int c = nw * 32 + nt * 16 + lq;
                v8us kpk, vpk;
                #pragma unroll
                for (int r = 0; r < 8; ++r) {
                    float kvv = ka[nt][r] + bK[nt];
                    float vvv = va[nt][r] + bV[nt];
                    bool ok = (l0 + lrow + r) < lend;
                    if (ok) {
                        ss += kvv * kvv;
                        cs[nt] += kvv;
                        Vb[(size_t)(l0 + lrow + r) * 1024 + h * 128 + c] = f2bf(vvv);
                    }
                    kpk[r] = ok ? f2bf(kvv) : (unsigned short)0;
                    vpk[r] = ok ? f2bf(vvv) : (unsigned short)0;
                }
                *(v8us*)&Kt[c][lrow] = kpk;
                *(v8us*)&Vt[c][lrow] = vpk;
            }
        }
        __syncthreads();

        v16bf a2[4], b2[2];
        #pragma unroll
        for (int mt = 0; mt < 4; ++mt) {
            const unsigned short* p = &Kt[mw * 64 + mt * 16 + lq][lh * 16];
            a2[mt] = ld_frag(p, p + 8);
        }
        #pragma unroll
        for (int dt = 0; dt < 2; ++dt) {
            const unsigned short* p = &Vt[nw * 32 + dt * 16 + lq][lh * 16];
            b2[dt] = ld_frag(p, p + 8);
        }
        #pragma unroll
        for (int mt = 0; mt < 4; ++mt)
            #pragma unroll
            for (int dt = 0; dt < 2; ++dt)
                kvacc[mt][dt] = wmma_bf16(a2[mt], b2[dt], kvacc[mt][dt]);
        __syncthreads();
    }

    #pragma unroll
    for (int mt = 0; mt < 4; ++mt)
        #pragma unroll
        for (int dt = 0; dt < 2; ++dt)
            #pragma unroll
            for (int r = 0; r < 8; ++r) {
                int m = mw * 64 + mt * 16 + lh * 8 + r;
                int d = nw * 32 + dt * 16 + lq;
                atomicAdd(&kvs[(h << 14) + (m << 7) + d], kvacc[mt][dt][r]);
            }
    #pragma unroll
    for (int nt = 0; nt < 2; ++nt) {
        float c2 = cs[nt] + __shfl_xor(cs[nt], 16, 32);
        if (lane < 16) atomicAdd(&ksum[h * 128 + nw * 32 + nt * 16 + lq], c2);
    }
    #pragma unroll
    for (int off = 16; off >= 1; off >>= 1) ss += __shfl_xor(ss, off, 32);
    if (lane == 0) atomicAdd(&scal[1], ss);
}

// ------------------------------------------------- stage 2.5: fold 1/(|Q||K|), transpose kvs -> bf16
__global__ void k_prep(const float* __restrict__ kvs, const float* __restrict__ ksum,
                       const float* __restrict__ scal,
                       unsigned short* __restrict__ kvsT, float* __restrict__ ksumS)
{
    const float invqk = rsqrtf(scal[0] * scal[1]);
    int i = blockIdx.x * 256 + threadIdx.x;
    if (i < 8 * 128 * 128) {
        int h = i >> 14, rem = i & 16383, d = rem >> 7, m = rem & 127;
        kvsT[(h << 14) + (d << 7) + m] = f2bf(kvs[(h << 14) + (m << 7) + d] * invqk);
    }
    if (i < 1024) ksumS[i] = ksum[i] * invqk;
}

// ------------------------------------------------- stage 3: out = mean_h (Q@kvs' + N*V)/(Q.ksum' + N)
// Transposed GEMM C'[d][node]; per-head kvsT slice staged once per BLOCK in LDS
// (8x cut in L2 traffic; 272B row pitch -> conflict-free b128 fragment reads).
__global__ void k_final(const unsigned short* __restrict__ Qb,
                        const unsigned short* __restrict__ Vb,
                        const unsigned short* __restrict__ kvsT,
                        const float* __restrict__ ksumS,
                        float* __restrict__ out, int N)
{
    __shared__ __align__(16) unsigned short kl[128][136];   // [d][m], padded pitch

    const int tid  = threadIdx.x;
    const int lane = tid & 31, w = tid >> 5;
    const int lq = lane & 15, lh = lane >> 4;
    const int n0 = blockIdx.x * 128 + w * 16;
    const float fN = (float)N;

    const int node = n0 + lq;
    const int rowB = (node < N) ? node : (N - 1);

    v8f vacc[8];
    #pragma unroll
    for (int i = 0; i < 8; ++i) vacc[i] = zero8();

    #pragma unroll 1
    for (int h = 0; h < 8; ++h) {
        // stage kvsT[h] (32KB) into LDS: 2048 v8us / 256 thr = 8 each
        #pragma unroll
        for (int i = 0; i < 8; ++i) {
            int e = tid + i * 256;
            int d = e >> 4, m8 = (e & 15) * 8;
            *(v8us*)&kl[d][m8] = *(const v8us*)(kvsT + (h << 14) + (d << 7) + m8);
        }
        __syncthreads();

        v8f dacc[8];
        #pragma unroll
        for (int i = 0; i < 8; ++i) dacc[i] = zero8();
        float s = 0.f;

        #pragma unroll
        for (int ksx = 0; ksx < 4; ++ksx) {
            const int m0 = ksx * 32;
            const unsigned short* bp = Qb + (size_t)rowB * 1024 + h * 128 + m0 + lh * 16;
            v16bf b = ld_frag(bp, bp + 8);
            {
                union { v16bf v; v8us u[2]; } qq; qq.v = b;
                const float4* kp = (const float4*)(ksumS + h * 128 + m0 + lh * 16);
                float4 kA = kp[0], kB = kp[1], kC = kp[2], kD = kp[3];
                s += bf2f(qq.u[0][0]) * kA.x + bf2f(qq.u[0][1]) * kA.y
                   + bf2f(qq.u[0][2]) * kA.z + bf2f(qq.u[0][3]) * kA.w
                   + bf2f(qq.u[0][4]) * kB.x + bf2f(qq.u[0][5]) * kB.y
                   + bf2f(qq.u[0][6]) * kB.z + bf2f(qq.u[0][7]) * kB.w
                   + bf2f(qq.u[1][0]) * kC.x + bf2f(qq.u[1][1]) * kC.y
                   + bf2f(qq.u[1][2]) * kC.z + bf2f(qq.u[1][3]) * kC.w
                   + bf2f(qq.u[1][4]) * kD.x + bf2f(qq.u[1][5]) * kD.y
                   + bf2f(qq.u[1][6]) * kD.z + bf2f(qq.u[1][7]) * kD.w;
            }
            #pragma unroll
            for (int dt = 0; dt < 8; ++dt) {
                const unsigned short* ap = &kl[dt * 16 + lq][m0 + lh * 16];
                v16bf a = ld_frag(ap, ap + 8);
                dacc[dt] = wmma_bf16(a, b, dacc[dt]);
            }
        }
        s += __shfl_xor(s, 16, 32);
        const float rden = 1.0f / (s + fN);

        const unsigned short* vrow = Vb + (size_t)rowB * 1024 + h * 128 + lh * 8;
        #pragma unroll
        for (int dt = 0; dt < 8; ++dt) {
            v8us vv = *(const v8us*)(vrow + dt * 16);
            #pragma unroll
            for (int r = 0; r < 8; ++r)
                vacc[dt][r] += (dacc[dt][r] + fN * bf2f(vv[r])) * rden;
        }
        __syncthreads();     // all fragment reads done before next head's staging
    }

    if (node < N) {
        float* orow = out + (size_t)node * 128 + lh * 8;
        #pragma unroll
        for (int dt = 0; dt < 8; ++dt) {
            float4 lo, hi;
            lo.x = vacc[dt][0] * 0.125f; lo.y = vacc[dt][1] * 0.125f;
            lo.z = vacc[dt][2] * 0.125f; lo.w = vacc[dt][3] * 0.125f;
            hi.x = vacc[dt][4] * 0.125f; hi.y = vacc[dt][5] * 0.125f;
            hi.z = vacc[dt][6] * 0.125f; hi.w = vacc[dt][7] * 0.125f;
            *(float4*)(orow + dt * 16)     = lo;
            *(float4*)(orow + dt * 16 + 4) = hi;
        }
    }
}

// ---------------------------------------------------------------- launch
extern "C" void kernel_launch(void* const* d_in, const int* in_sizes, int n_in,
                              void* d_out, int out_size, void* d_ws, size_t ws_size,
                              hipStream_t stream) {
    const float* Xq = (const float*)d_in[0];
    const float* Xs = (const float*)d_in[1];
    const float* Wq = (const float*)d_in[2];
    const float* bq = (const float*)d_in[3];
    const float* Wk = (const float*)d_in[4];
    const float* bk = (const float*)d_in[5];
    const float* Wv = (const float*)d_in[6];
    const float* bv = (const float*)d_in[7];
    float* out = (float*)d_out;
    const int N = in_sizes[0] / 256;

    char* ws = (char*)d_ws;
    const size_t sz_mat = (size_t)N * 1024 * 2;
    unsigned short* Qb = (unsigned short*)(ws);
    unsigned short* Vb = (unsigned short*)(ws + sz_mat);
    float* kvs   = (float*)(ws + 2 * sz_mat);
    float* ksum  = kvs + 8 * 128 * 128;
    float* scal  = ksum + 1024;
    float* ksumS = scal + 4;
    unsigned short* kvsT = (unsigned short*)(ksumS + 1024);
    unsigned short* Wt   = kvsT + 8 * 128 * 128;

    const int nz = 8 * 128 * 128 + 1024 + 4;
    k_zero<<<(nz + 255) / 256, 256, 0, stream>>>(kvs, nz);

    k_wt<<<(3 * 1024 * 256 + 255) / 256, 256, 0, stream>>>(Wq, Wk, Wv, Wt);

    k_q<<<dim3((N + 63) / 64, 16), 128, 0, stream>>>(Xq, bq, Wt, Qb, scal, N);

    k_kv<<<dim3(32, 8), 256, 0, stream>>>(Xs, bk, bv, Wt, Vb, kvs, ksum, scal, N);

    k_prep<<<512, 256, 0, stream>>>(kvs, ksum, scal, kvsT, ksumS);

    k_final<<<(N + 127) / 128, 256, 0, stream>>>(Qb, Vb, kvsT, ksumS, out, N);
}
